// ImprovedAdaptiveSystem_20770461844032
// MI455X (gfx1250) — compile-verified, matrix operand order checked by reading, not run on hardware
//
#include <hip/hip_runtime.h>
#include <hip/hip_bf16.h>

typedef __attribute__((ext_vector_type(16))) _Float16 v16h;
typedef __attribute__((ext_vector_type(8)))  float    v8f;

#define IMG   128
#define CIN   3
#define HIN   224
#define WIN   224
#define OCH   64              // channels per expert
#define NCH   128             // combined (t ++ f)
#define HOUT  112
#define WOUT  112
#define KPAD  32              // K = 27 weights + 1 bias slot (k=27), padded to 32
#define NPIX  (HOUT * WOUT)   // 12544

// ---------------- workspace layout ----------------
// [0)        float  acc[IMG][NCH]          65536 B
// [65536)    f16    wpack[NCH][KPAD]        8192 B
#define WS_ACC_OFF   0
#define WS_WPK_OFF   65536

// Pack both experts' conv weights (and bias, folded into the K=27 GEMM slot)
// into f16 B-matrix form, and zero the pooling accumulator (every launch:
// graph replays don't re-poison the workspace).
__global__ void prep_kernel(const float* __restrict__ tw, const float* __restrict__ tb,
                            const float* __restrict__ fw, const float* __restrict__ fb,
                            float* __restrict__ acc, _Float16* __restrict__ wpack)
{
    int tid = blockIdx.x * blockDim.x + threadIdx.x;
    int nthreads = gridDim.x * blockDim.x;
    for (int i = tid; i < IMG * NCH; i += nthreads) acc[i] = 0.0f;
    for (int i = tid; i < NCH * KPAD; i += nthreads) {
        int n = i / KPAD, k = i % KPAD;
        float v = 0.0f;
        if (k < 27) {                               // (O,C,3,3) contiguous: k = c*9 + r*3 + s
            v = ((n < OCH) ? tw : fw)[(n & 63) * 27 + k];
        } else if (k == 27) {                       // bias folded into GEMM (A supplies 1.0)
            v = (n < OCH) ? tb[n] : fb[n - OCH];
        }
        wpack[i] = (_Float16)v;
    }
}

// Implicit-GEMM conv (stride 2, SAME) + bias(folded) + ReLU + global-average
// partial sums. One wave handles 16-pixel row tiles; 8 WMMAs/tile = 128 ch.
__global__ __launch_bounds__(256)
void conv_pool_kernel(const float* __restrict__ x, const _Float16* __restrict__ wpack,
                      float* __restrict__ acc)
{
    const int lane  = threadIdx.x & 31;
    const int wave  = threadIdx.x >> 5;
    const int img   = blockIdx.y;
    const int laneN = lane & 15;
    const int hi8   = (lane & 16) ? 8 : 0;   // lanes 16..31 hold the K+8 half (ISA A/B layout)

    // B fragments: weights+bias for all 8 N-tiles, resident in VGPRs for the wave.
    v16h bfr[8];
    #pragma unroll
    for (int t = 0; t < 8; ++t) {
        const int n = t * 16 + laneN;
        #pragma unroll
        for (int i = 0; i < 16; ++i) {
            const int khalf = i + (i >= 8 ? 8 : 0);    // K = 0..7,16..23 (+8 on upper lanes)
            bfr[t][i] = wpack[n * KPAD + khalf + hi8]; // k>=28 slots are pre-zeroed
        }
    }

    // Per-lane patch gather offsets (fixed for the wave): k -> (c,r,s).
    // k==27 is the bias slot: A supplies a constant 1.0 there.
    int   offs[16], rr[16], ss[16];
    float fill[16];
    #pragma unroll
    for (int i = 0; i < 16; ++i) {
        const int khalf = i + (i >= 8 ? 8 : 0);
        const int k = khalf + hi8;
        if (k < 27) {
            const int c = k / 9, rem = k % 9, r = rem / 3, s = rem % 3;
            offs[i] = c * (HIN * WIN) + r * WIN + s;
            rr[i] = r; ss[i] = s; fill[i] = 0.0f;
        } else {
            offs[i] = -1; rr[i] = 0; ss[i] = 0;
            fill[i] = (k == 27) ? 1.0f : 0.0f;         // bias multiplier
        }
    }

    const float* xb = x + (size_t)img * (CIN * HIN * WIN);
    float part[8];
    #pragma unroll
    for (int t = 0; t < 8; ++t) part[t] = 0.0f;

    const int tilesPerImg = HOUT * (WOUT / 16);            // 112 * 7 = 784
    const int wavesPerBlk = blockDim.x >> 5;
    const int wstride     = gridDim.x * wavesPerBlk;

    for (int tIdx = blockIdx.x * wavesPerBlk + wave; tIdx < tilesPerImg; tIdx += wstride) {
        const int oy  = tIdx / (WOUT / 16);
        const int ox  = (tIdx % (WOUT / 16)) * 16 + laneN; // this lane's pixel (row M = laneN)
        const int iy0 = oy * 2, ix0 = ox * 2;              // SAME pad: 0 top/left, 1 bottom/right
        const float* p = xb + iy0 * WIN + ix0;

        v16h a;
        #pragma unroll
        for (int i = 0; i < 16; ++i) {
            float v = fill[i];
            if (offs[i] >= 0 && (iy0 + rr[i]) < HIN && (ix0 + ss[i]) < WIN)
                v = p[offs[i]];
            a[i] = (_Float16)v;
        }

        #pragma unroll
        for (int t = 0; t < 8; ++t) {
            v8f c = {};
            c = __builtin_amdgcn_wmma_f32_16x16x32_f16(
                    /*neg_a=*/false, a, /*neg_b=*/false, bfr[t],
                    /*c_mod=*/(short)0, c, /*reuse_a=*/false, /*reuse_b=*/false);
            float s = 0.0f;
            #pragma unroll
            for (int v2 = 0; v2 < 8; ++v2)                 // 8 pixel rows held by this lane
                s += fmaxf(c[v2], 0.0f);                   // ReLU -> single v_max_num_f32
            part[t] += s;
        }
    }

    // Merge the two M-halves (lanes L and L+16 hold the same channel), then one
    // float atomic per (image, channel) per wave.
    #pragma unroll
    for (int t = 0; t < 8; ++t) {
        const float v = part[t] + __shfl_xor(part[t], 16, 32);
        if (lane < 16)
            atomicAdd(&acc[img * NCH + t * 16 + laneN], v);
    }
}

// mean -> FC -> softmax confidence -> threshold blend. 128 images, 1 thread each.
__global__ void head_kernel(const float* __restrict__ acc,
                            const float* __restrict__ t_fc_w, const float* __restrict__ t_fc_b,
                            const float* __restrict__ f_fc_w, const float* __restrict__ f_fc_b,
                            float* __restrict__ out)
{
    const int img = blockIdx.x * blockDim.x + threadIdx.x;
    if (img >= IMG) return;
    const float inv = 1.0f / (float)NPIX;
    float t0 = t_fc_b[0], t1 = t_fc_b[1], f0 = f_fc_b[0], f1 = f_fc_b[1];
    #pragma unroll 8
    for (int o = 0; o < OCH; ++o) {
        const float gt = acc[img * NCH + o] * inv;
        const float gf = acc[img * NCH + OCH + o] * inv;
        t0 += gt * t_fc_w[o];
        t1 += gt * t_fc_w[OCH + o];
        f0 += gf * f_fc_w[o];
        f1 += gf * f_fc_w[OCH + o];
    }
    const float m  = fmaxf(t0, t1);
    const float e0 = __expf(t0 - m), e1 = __expf(t1 - m);
    const float conf = fmaxf(e0, e1) / (e0 + e1);
    float o0, o1;
    if (conf < 0.4f)       { o0 = f0; o1 = f1; }
    else if (conf < 0.55f) { o0 = 0.3f * t0 + 0.7f * f0; o1 = 0.3f * t1 + 0.7f * f1; }
    else                   { o0 = t0; o1 = t1; }
    out[img * 2 + 0] = o0;
    out[img * 2 + 1] = o1;
}

extern "C" void kernel_launch(void* const* d_in, const int* in_sizes, int n_in,
                              void* d_out, int out_size, void* d_ws, size_t ws_size,
                              hipStream_t stream) {
    const float* x       = (const float*)d_in[0];
    const float* t_cw    = (const float*)d_in[1];
    const float* t_cb    = (const float*)d_in[2];
    const float* t_fw    = (const float*)d_in[3];
    const float* t_fb    = (const float*)d_in[4];
    const float* f_cw    = (const float*)d_in[5];
    const float* f_cb    = (const float*)d_in[6];
    const float* f_fw    = (const float*)d_in[7];
    const float* f_fb    = (const float*)d_in[8];

    char* ws = (char*)d_ws;
    float*     acc   = (float*)(ws + WS_ACC_OFF);
    _Float16*  wpack = (_Float16*)(ws + WS_WPK_OFF);
    float*     out   = (float*)d_out;

    prep_kernel<<<dim3(64), dim3(256), 0, stream>>>(t_cw, t_cb, f_cw, f_cb, acc, wpack);
    conv_pool_kernel<<<dim3(16, IMG), dim3(256), 0, stream>>>(x, wpack, acc);
    head_kernel<<<dim3(1), dim3(128), 0, stream>>>(acc, t_fw, t_fb, f_fw, f_fb, out);
}
